// DeltaAttention_2156073582758
// MI455X (gfx1250) — matrix operand order repaired and checked by reading, not compiled, not run on hardware
//
#include <hip/hip_runtime.h>
#include <math.h>

// ---------------------------------------------------------------------------
// DeltaAttention forward for MI455X (gfx1250, wave32, WMMA + async->LDS)
// B=4, L=2048, D=1024, H=16, HD=64
// ---------------------------------------------------------------------------

typedef __attribute__((ext_vector_type(16))) _Float16 v16h;
typedef __attribute__((ext_vector_type(8)))  float    v8f;

#define BB   4
#define LL   2048
#define DD   1024
#define HH   16
#define HD   64
#define ML   (BB*LL)        // 8192 rows

// ---- CDNA5 async global->LDS copy (ASYNCcnt-tracked), inline asm ----------
__device__ __forceinline__ void async_ld_b128(unsigned int lds_off,
                                              const void* gptr) {
    unsigned long long ga = (unsigned long long)(uintptr_t)gptr;
    asm volatile("global_load_async_to_lds_b128 %0, %1, off"
                 :: "v"(lds_off), "v"(ga) : "memory");
}
__device__ __forceinline__ void wait_async_le4() {
    asm volatile("s_wait_asynccnt 0x4" ::: "memory");
}
__device__ __forceinline__ void wait_async_0() {
    asm volatile("s_wait_asynccnt 0x0" ::: "memory");
}

// ------------------------------ f32 -> f16 convert -------------------------
__global__ void cvt_f32_f16(const float* __restrict__ in,
                            _Float16* __restrict__ out, int n) {
    int i = blockIdx.x * blockDim.x + threadIdx.x;
    int stride = gridDim.x * blockDim.x;
    for (; i < n; i += stride) out[i] = (_Float16)in[i];
}

// ------------------------------ WMMA GEMM ----------------------------------
// C[M,N] = A[M,K] * W[N,K]^T.  A,W f16 row-major, C f32 (leading dim ldc).
// Block: 256 threads (8 waves). Block tile 128x128, K step 32.
// Wave (wm 0..3, wn 0..1) computes 32x64 = 2x4 WMMA frags (8 wmma / K-step),
// LDS double-buffered and filled by global_load_async_to_lds_b128.
#define BM 128
#define BN 128
#define BK 32
#define LDAS 40   // padded LDS row stride (halves)
#define LDBS 40

union Frag { v16h v; unsigned int u[8]; };

__global__ __launch_bounds__(256) void wmma_gemm_f16(
    const _Float16* __restrict__ A, const _Float16* __restrict__ W,
    float* __restrict__ C, int M, int N, int K, int ldc)
{
    __shared__ _Float16 As[2][BM * LDAS];
    __shared__ _Float16 Bs[2][BN * LDBS];

    const int tid  = threadIdx.x;
    const int wid  = tid >> 5;
    const int lane = tid & 31;
    const int wm   = wid & 3;       // 0..3 -> 32-row band
    const int wn   = wid >> 2;      // 0..1 -> 64-col band
    const int lm   = lane & 15;
    const int lh   = lane >> 4;

    const int m0 = blockIdx.x * BM;     // blockIdx.x = M tiles (more blocks)
    const int n0 = blockIdx.y * BN;

    const int grow = tid >> 2;      // 0..63 loader row
    const int gq   = tid & 3;       // 16B quad within a 32-half row

    v8f acc[8] = {};

    // per-thread async-issue of one K-tile into LDS stage s (4 x b128)
    auto issue = [&](int k0, int s) {
        async_ld_b128((unsigned int)(uintptr_t)&As[s][grow * LDAS + gq * 8],
                      A + (size_t)(m0 + grow) * K + k0 + gq * 8);
        async_ld_b128((unsigned int)(uintptr_t)&As[s][(64 + grow) * LDAS + gq * 8],
                      A + (size_t)(m0 + 64 + grow) * K + k0 + gq * 8);
        async_ld_b128((unsigned int)(uintptr_t)&Bs[s][grow * LDBS + gq * 8],
                      W + (size_t)(n0 + grow) * K + k0 + gq * 8);
        async_ld_b128((unsigned int)(uintptr_t)&Bs[s][(64 + grow) * LDBS + gq * 8],
                      W + (size_t)(n0 + 64 + grow) * K + k0 + gq * 8);
    };

    const int nk = K / BK;
    issue(0, 0);

    for (int kt = 0; kt < nk; kt++) {
        const int cur = kt & 1;
        if (kt + 1 < nk) {
            issue((kt + 1) * BK, cur ^ 1);   // overlap next tile with compute
            wait_async_le4();                // current tile's 4 copies landed
        } else {
            wait_async_0();
        }
        __syncthreads();                     // LDS visible to all waves

        // ---- load fragments per ISA 16-bit layouts
        Frag af[2], bf[4];
        #pragma unroll
        for (int sm = 0; sm < 2; sm++) {
            const unsigned int* rp = reinterpret_cast<const unsigned int*>(
                &As[cur][(wm * 32 + sm * 16 + lm) * LDAS]);
            // A 16x32: lanes 0-15 K=0..7,16..23 ; lanes 16-31 K=8..15,24..31
            #pragma unroll
            for (int r = 0; r < 4; r++) af[sm].u[r]     = rp[lh * 4 + r];
            #pragma unroll
            for (int r = 0; r < 4; r++) af[sm].u[4 + r] = rp[8 + lh * 4 + r];
        }
        #pragma unroll
        for (int sn = 0; sn < 4; sn++) {
            const unsigned int* rp = reinterpret_cast<const unsigned int*>(
                &Bs[cur][(wn * 64 + sn * 16 + lm) * LDBS]);
            // B 32x16: lanes 0-15 K=0..7,16..23 ; lanes 16-31 K=8..15,24..31
            #pragma unroll
            for (int r = 0; r < 4; r++) bf[sn].u[r]     = rp[lh * 4 + r];
            #pragma unroll
            for (int r = 0; r < 4; r++) bf[sn].u[4 + r] = rp[8 + lh * 4 + r];
        }

        // ---- 8 WMMA ops per wave per K-step (A reused x4, B reused x2)
        #pragma unroll
        for (int sm = 0; sm < 2; sm++)
            #pragma unroll
            for (int sn = 0; sn < 4; sn++)
                acc[sm * 4 + sn] = __builtin_amdgcn_wmma_f32_16x16x32_f16(
                    false, af[sm].v, false, bf[sn].v,
                    (short)0, acc[sm * 4 + sn], false, false);

        __syncthreads();  // all reads done before this stage is refilled
    }

    // ---- store C per ISA 32-bit C/D layout: lane n=lm, rows lh*8+r
    #pragma unroll
    for (int sm = 0; sm < 2; sm++)
        #pragma unroll
        for (int sn = 0; sn < 4; sn++)
            #pragma unroll
            for (int r = 0; r < 8; r++) {
                int row = m0 + wm * 32 + sm * 16 + lh * 8 + r;
                int col = n0 + wn * 64 + sn * 16 + lm;
                C[(size_t)row * ldc + col] = acc[sm * 4 + sn][r];
            }
}

// ------------------------- causal depthwise conv (q) -----------------------
__global__ void conv_q_kernel(const float* __restrict__ qkv,
                              const float* __restrict__ qw,
                              const float* __restrict__ qb,
                              float* __restrict__ qc)
{
    int idx = blockIdx.x * blockDim.x + threadIdx.x;
    int stride = gridDim.x * blockDim.x;
    for (; idx < ML * DD; idx += stride) {
        int d = idx & (DD - 1);
        int l = (idx >> 10) & (LL - 1);
        int b = idx >> 21;
        float acc = qb[d];
        #pragma unroll
        for (int j = 0; j < 4; j++) {
            int t = l - 3 + j;
            if (t >= 0)
                acc += qw[d * 4 + j] * qkv[((size_t)(b * LL + t)) * 3072 + d];
        }
        qc[idx] = acc;
    }
}

// ------------------- conv (k) + per-head L2 normalize ----------------------
__global__ __launch_bounds__(64) void conv_k_norm_kernel(
    const float* __restrict__ qkv, const float* __restrict__ kw,
    const float* __restrict__ kb, float* __restrict__ kc)
{
    __shared__ float ss[64];
    int bl = blockIdx.x >> 4;         // b*L + l
    int h  = blockIdx.x & 15;
    int l  = bl & (LL - 1);
    int i  = threadIdx.x;
    int d  = h * HD + i;

    float val = kb[d];
    #pragma unroll
    for (int j = 0; j < 4; j++) {
        int t = l - 3 + j;
        if (t >= 0)
            val += kw[d * 4 + j] * qkv[((size_t)(bl - l + t)) * 3072 + 1024 + d];
    }
    ss[i] = val * val;
    __syncthreads();
    float s = 0.f;
    #pragma unroll
    for (int j = 0; j < 64; j++) s += ss[j];
    kc[(size_t)bl * DD + d] = val / fmaxf(sqrtf(s), 1e-12f);
}

// ------------------------------ beta = sigmoid(x Wb^T + bb) ----------------
__global__ __launch_bounds__(256) void beta_kernel(
    const float* __restrict__ x, const float* __restrict__ Wb,
    const float* __restrict__ bbv, float* __restrict__ betab)
{
    __shared__ float red[256];
    int bl   = blockIdx.x;            // b*L + l
    int h    = threadIdx.x & 15;
    int slot = threadIdx.x >> 4;
    float a = 0.f;
    for (int k = slot; k < DD; k += 16)
        a += x[(size_t)bl * DD + k] * Wb[(size_t)h * DD + k];
    red[slot * 16 + h] = a;
    __syncthreads();
    if (slot == 0) {
        float s = bbv[h];
        #pragma unroll
        for (int j = 0; j < 16; j++) s += red[j * 16 + h];
        betab[(size_t)bl * HH + h] = 1.f / (1.f + expf(-s));
    }
}

// ------------------------------ delta-rule scan ----------------------------
// One workgroup (64 lanes) per (b,h). Lane i owns row i of S in registers.
// q/k/v/beta for step t+1 are prefetched into registers during step t.
__global__ __launch_bounds__(64) void scan_kernel(
    const float* __restrict__ qc, const float* __restrict__ kc,
    const float* __restrict__ qkv, const float* __restrict__ betab,
    _Float16* __restrict__ yh, float* __restrict__ S_out,
    float* __restrict__ z_out)
{
    const int bh = blockIdx.x;        // 0..63
    const int b  = bh >> 4;
    const int h  = bh & 15;
    const int i  = threadIdx.x;       // 0..63 (row of S)

    __shared__ float qs[64], ks[64], vs[64], zs[64];

    float S[64];
    #pragma unroll
    for (int j = 0; j < 64; j++) S[j] = 0.f;
    float zi = 0.f;

    const size_t baseD = (size_t)b * LL * DD + (size_t)h * HD;
    const size_t base3 = (size_t)b * LL * 3072 + 2048 + (size_t)h * HD;
    const size_t baseB = (size_t)b * LL * HH + h;

    // prefetch t = 0
    float qn = qc[baseD + i];
    float kn = kc[baseD + i];
    float vn = qkv[base3 + i];
    float bn = betab[baseB];

    for (int t = 0; t < LL; t++) {
        qs[i] = qn; ks[i] = kn; vs[i] = vn; zs[i] = zi;
        float bt = bn;
        __syncthreads();

        if (t + 1 < LL) {   // issue next-step loads; consumed after barrier
            qn = qc[baseD + (size_t)(t + 1) * DD + i];
            kn = kc[baseD + (size_t)(t + 1) * DD + i];
            vn = qkv[base3 + (size_t)(t + 1) * 3072 + i];
            bn = betab[baseB + (size_t)(t + 1) * HH];
        }

        float num = 0.f, old = 0.f, den = 0.f;
        #pragma unroll
        for (int j = 0; j < 64; j++) {
            num += S[j] * qs[j];
            old += S[j] * ks[j];
            den += zs[j] * qs[j];
        }
        float y    = num / (den + 1e-6f);
        float coef = bt * (vs[i] - old);
        #pragma unroll
        for (int j = 0; j < 64; j++) S[j] += coef * ks[j];
        zi += bt * ks[i];

        yh[baseD + (size_t)t * DD + i] = (_Float16)y;
        __syncthreads();
    }

    #pragma unroll
    for (int j = 0; j < 64; j++)
        S_out[((size_t)bh * 64 + i) * 64 + j] = S[j];
    z_out[(size_t)bh * 64 + i] = zi;
}

// ------------------------------ LayerNorm ----------------------------------
__global__ __launch_bounds__(256) void ln_kernel(
    const float* __restrict__ preln, const float* __restrict__ lnw,
    const float* __restrict__ lnb, float* __restrict__ out)
{
    __shared__ float s1[256], s2[256];
    __shared__ float mu_s, rstd_s;
    int bl  = blockIdx.x;
    int tid = threadIdx.x;
    const float* row = preln + (size_t)bl * DD;

    float sum = 0.f, ssq = 0.f;
    float v[4];
    #pragma unroll
    for (int r = 0; r < 4; r++) {
        v[r] = row[tid * 4 + r];
        sum += v[r];
        ssq += v[r] * v[r];
    }
    s1[tid] = sum; s2[tid] = ssq;
    __syncthreads();
    for (int st = 128; st > 0; st >>= 1) {
        if (tid < st) { s1[tid] += s1[tid + st]; s2[tid] += s2[tid + st]; }
        __syncthreads();
    }
    if (tid == 0) {
        float mu  = s1[0] * (1.f / DD);
        float var = s2[0] * (1.f / DD) - mu * mu;
        mu_s = mu; rstd_s = rsqrtf(var + 1e-5f);
    }
    __syncthreads();
    float mu = mu_s, rstd = rstd_s;
    #pragma unroll
    for (int r = 0; r < 4; r++) {
        int d = tid * 4 + r;
        out[(size_t)bl * DD + d] = (v[r] - mu) * rstd * lnw[d] + lnb[d];
    }
}

// ------------------------------ launch -------------------------------------
extern "C" void kernel_launch(void* const* d_in, const int* in_sizes, int n_in,
                              void* d_out, int out_size, void* d_ws, size_t ws_size,
                              hipStream_t stream) {
    const float* x   = (const float*)d_in[0];
    const float* Wq  = (const float*)d_in[1];
    const float* Wk  = (const float*)d_in[2];
    const float* Wv  = (const float*)d_in[3];
    const float* Wo  = (const float*)d_in[4];
    const float* Wb  = (const float*)d_in[5];
    const float* bbv = (const float*)d_in[6];
    const float* cqw = (const float*)d_in[7];
    const float* cqb = (const float*)d_in[8];
    const float* ckw = (const float*)d_in[9];
    const float* ckb = (const float*)d_in[10];
    const float* lnw = (const float*)d_in[11];
    const float* lnb = (const float*)d_in[12];
    float* out = (float*)d_out;

    // workspace layout (bytes)
    char* ws = (char*)d_ws;
    _Float16* xh    = (_Float16*)(ws);                       // 8192x1024 f16
    _Float16* wh    = (_Float16*)(ws + 16777216);            // 4096x1024 f16 (Wq|Wk|Wv|Wo)
    float*    qkv   = (float*)   (ws + 25165824);            // 8192x3072 f32
    float*    qc    = (float*)   (ws + 125829120);           // 8192x1024 f32
    float*    kc    = (float*)   (ws + 159383552);           // 8192x1024 f32
    float*    betab = (float*)   (ws + 192937984);           // 8192x16 f32
    _Float16* yh    = (_Float16*)(ws + 193462272);           // 8192x1024 f16
    float*    preln = (float*)   (ws + 210239488);           // 8192x1024 f32

    float* S_out = out + (size_t)ML * DD;                    // B*H*64*64
    float* z_out = S_out + (size_t)BB * HH * HD * HD;        // B*H*64

    // 1) convert x + pack weights to f16
    cvt_f32_f16<<<4096, 256, 0, stream>>>(x,  xh, ML * DD);
    cvt_f32_f16<<<1024, 256, 0, stream>>>(Wq, wh + 0 * 1048576, DD * DD);
    cvt_f32_f16<<<1024, 256, 0, stream>>>(Wk, wh + 1 * 1048576, DD * DD);
    cvt_f32_f16<<<1024, 256, 0, stream>>>(Wv, wh + 2 * 1048576, DD * DD);
    cvt_f32_f16<<<1024, 256, 0, stream>>>(Wo, wh + 3 * 1048576, DD * DD);

    // 2) fused qkv projection: [8192,1024] x [3072,1024]^T -> [8192,3072]
    wmma_gemm_f16<<<dim3(ML / BM, 3072 / BN), 256, 0, stream>>>(
        xh, wh, qkv, ML, 3072, DD, 3072);

    // 3) causal depthwise convs + k-normalize + beta gate
    conv_q_kernel<<<8192, 256, 0, stream>>>(qkv, cqw, cqb, qc);
    conv_k_norm_kernel<<<ML * HH, 64, 0, stream>>>(qkv, ckw, ckb, kc);
    beta_kernel<<<ML, 256, 0, stream>>>(x, Wb, bbv, betab);

    // 4) sequential delta-rule scan (one WG of 64 lanes per (b,h))
    scan_kernel<<<BB * HH, 64, 0, stream>>>(qc, kc, qkv, betab, yh, S_out, z_out);

    // 5) output projection: [8192,1024] x [1024,1024]^T -> [8192,1024]
    wmma_gemm_f16<<<dim3(ML / BM, DD / BN), 256, 0, stream>>>(
        yh, wh + 3 * 1048576, preln, ML, DD, DD, DD);

    // 6) LayerNorm -> d_out
    ln_kernel<<<ML, 256, 0, stream>>>(preln, lnw, lnb, out);
}
